// QuantizedPCMEProjector_12446815224108
// MI455X (gfx1250) — 8-bit matrix operand order repaired and checked by reading, compiled, not run on hardware
//
#include <hip/hip_runtime.h>
#include <hip/hip_bf16.h>
#include <math.h>


typedef __attribute__((ext_vector_type(8))) int v8i;
typedef __attribute__((ext_vector_type(4))) int v4i;
typedef __attribute__((ext_vector_type(2))) int v2i;

#define LDS_STRIDE 80

#if __has_builtin(__builtin_amdgcn_global_load_async_to_lds_b128)
#define HAVE_ASYNC_LDS 1
#else
#define HAVE_ASYNC_LDS 0
#endif

#if HAVE_ASYNC_LDS
typedef __attribute__((address_space(1))) v4i g1v4i;
typedef __attribute__((address_space(3))) v4i l3v4i;

__device__ __forceinline__ void async_copy16(const signed char* g, unsigned char* l)
{
    __builtin_amdgcn_global_load_async_to_lds_b128(
        (g1v4i*)g, (l3v4i*)l,  0,  0);
}
__device__ __forceinline__ void wait_async0()
{
#if __has_builtin(__builtin_amdgcn_s_wait_asynccnt)
    __builtin_amdgcn_s_wait_asynccnt(0);
#else
    asm volatile("s_wait_asynccnt 0x0" ::: "memory");
#endif
}
#endif

__global__ __launch_bounds__(256)
void quantize_x_kernel(const float* __restrict__ x, signed char* __restrict__ xq, int n4)
{
    int i = blockIdx.x * blockDim.x + threadIdx.x;
    if (i >= n4) return;
    float4 v = ((const float4*)x)[i];
    float t0 = fminf(fmaxf(rintf(v.x * 50.0f), -128.0f), 127.0f);
    float t1 = fminf(fmaxf(rintf(v.y * 50.0f), -128.0f), 127.0f);
    float t2 = fminf(fmaxf(rintf(v.z * 50.0f), -128.0f), 127.0f);
    float t3 = fminf(fmaxf(rintf(v.w * 50.0f), -128.0f), 127.0f);
    int a = (int)t0, b = (int)t1, c = (int)t2, d = (int)t3;
    unsigned int p = (a & 255) | ((b & 255) << 8) | ((c & 255) << 16) | ((d & 255) << 24);
    ((unsigned int*)xq)[i] = p;
}

__global__ __launch_bounds__(256)
void pack_w_kernel(const int* __restrict__ w, signed char* __restrict__ wq, int n4)
{
    int i = blockIdx.x * blockDim.x + threadIdx.x;
    if (i >= n4) return;
    v4i v = ((const v4i*)w)[i];
    unsigned int p = (v[0] & 255) | ((v[1] & 255) << 8) | ((v[2] & 255) << 16) | ((v[3] & 255) << 24);
    ((unsigned int*)wq)[i] = p;
}

template<int MODE>
__global__ __launch_bounds__(256)
void gemm_i8_kernel(const signed char* __restrict__ A,
                    const signed char* __restrict__ W,
                    const int*         __restrict__ bias,
                    signed char*       __restrict__ hout,
                    float*             __restrict__ fout,
                    const float*       __restrict__ xres,
                    int M, int N, int K)
{
    __shared__ __align__(16) unsigned char xs[2][128 * LDS_STRIDE];
    __shared__ __align__(16) unsigned char ws[2][128 * LDS_STRIDE];

    const int tid   = threadIdx.x;
    const int lane  = tid & 31;
    const int wave  = tid >> 5;
    const int m_blk = blockIdx.x * 128;
    const int n_blk = blockIdx.y * 128;

    const int wm = (wave & 3) * 32;
    const int wn = (wave >> 2) * 64;

    const int c0 = tid, c1 = tid + 256;
    const int r0 = c0 >> 2, o0 = (c0 & 3) * 16;
    const int r1 = c1 >> 2, o1 = (c1 & 3) * 16;

    v8i acc[2][4];
    {
        v8i z = {};
        #pragma unroll
        for (int i = 0; i < 2; ++i)
            #pragma unroll
            for (int j = 0; j < 4; ++j) acc[i][j] = z;
    }

    const int ksteps = K >> 6;
    int buf = 0;

#if HAVE_ASYNC_LDS
    async_copy16(A + (size_t)(m_blk + r0) * K + o0, &xs[0][r0 * LDS_STRIDE + o0]);
    async_copy16(W + (size_t)(n_blk + r0) * K + o0, &ws[0][r0 * LDS_STRIDE + o0]);
    async_copy16(A + (size_t)(m_blk + r1) * K + o1, &xs[0][r1 * LDS_STRIDE + o1]);
    async_copy16(W + (size_t)(n_blk + r1) * K + o1, &ws[0][r1 * LDS_STRIDE + o1]);
    wait_async0();
    __syncthreads();
#else
    v4i xreg0, xreg1, wreg0, wreg1;
    xreg0 = *(const v4i*)(A + (size_t)(m_blk + r0) * K + o0);
    wreg0 = *(const v4i*)(W + (size_t)(n_blk + r0) * K + o0);
    xreg1 = *(const v4i*)(A + (size_t)(m_blk + r1) * K + o1);
    wreg1 = *(const v4i*)(W + (size_t)(n_blk + r1) * K + o1);
    *(v4i*)(&xs[0][r0 * LDS_STRIDE + o0]) = xreg0;
    *(v4i*)(&ws[0][r0 * LDS_STRIDE + o0]) = wreg0;
    *(v4i*)(&xs[0][r1 * LDS_STRIDE + o1]) = xreg1;
    *(v4i*)(&ws[0][r1 * LDS_STRIDE + o1]) = wreg1;
    __syncthreads();
#endif

    const int lr = lane & 15;
    const int lh = lane >> 4;

    for (int ks = 0; ks < ksteps; ++ks) {
#if HAVE_ASYNC_LDS
        if (ks + 1 < ksteps) {
            const int k0 = (ks + 1) << 6;
            const int nb = buf ^ 1;
            async_copy16(A + (size_t)(m_blk + r0) * K + k0 + o0, &xs[nb][r0 * LDS_STRIDE + o0]);
            async_copy16(W + (size_t)(n_blk + r0) * K + k0 + o0, &ws[nb][r0 * LDS_STRIDE + o0]);
            async_copy16(A + (size_t)(m_blk + r1) * K + k0 + o1, &xs[nb][r1 * LDS_STRIDE + o1]);
            async_copy16(W + (size_t)(n_blk + r1) * K + k0 + o1, &ws[nb][r1 * LDS_STRIDE + o1]);
        }
#else
        if (ks + 1 < ksteps) {
            const int k0 = (ks + 1) << 6;
            xreg0 = *(const v4i*)(A + (size_t)(m_blk + r0) * K + k0 + o0);
            wreg0 = *(const v4i*)(W + (size_t)(n_blk + r0) * K + k0 + o0);
            xreg1 = *(const v4i*)(A + (size_t)(m_blk + r1) * K + k0 + o1);
            wreg1 = *(const v4i*)(W + (size_t)(n_blk + r1) * K + k0 + o1);
        }
#endif

        const unsigned char* xb = xs[buf];
        const unsigned char* wb = ws[buf];

        v8i afrag[2];
        #pragma unroll
        for (int i = 0; i < 2; ++i) {
            const unsigned char* p = xb + (wm + i * 16 + lr) * LDS_STRIDE + lh * 16;
            v2i t0 = *(const v2i*)(p +  0);
            v2i t1 = *(const v2i*)(p +  8);
            v2i t2 = *(const v2i*)(p + 32);
            v2i t3 = *(const v2i*)(p + 40);
            v8i a;
            a[0] = t0[0]; a[1] = t0[1]; a[2] = t1[0]; a[3] = t1[1];
            a[4] = t2[0]; a[5] = t2[1]; a[6] = t3[0]; a[7] = t3[1];
            afrag[i] = a;
        }
        v8i bfrag[4];
        #pragma unroll
        for (int j = 0; j < 4; ++j) {
            const unsigned char* p = wb + (wn + j * 16 + lr) * LDS_STRIDE + lh * 16;
            v4i t0 = *(const v4i*)(p +  0);
            v4i t1 = *(const v4i*)(p + 32);
            v8i b;
            b[0] = t0[0]; b[1] = t0[1]; b[2] = t0[2]; b[3] = t0[3];
            b[4] = t1[0]; b[5] = t1[1]; b[6] = t1[2]; b[7] = t1[3];
            bfrag[j] = b;
        }

        #pragma unroll
        for (int i = 0; i < 2; ++i)
            #pragma unroll
            for (int j = 0; j < 4; ++j)
                acc[i][j] = __builtin_amdgcn_wmma_i32_16x16x64_iu8(
                     true, afrag[i],  true, bfrag[j],
                    acc[i][j],  false,  false);

#if HAVE_ASYNC_LDS
        if (ks + 1 < ksteps) {
            wait_async0();
            __syncthreads();
            buf ^= 1;
        }
#else
        __syncthreads();
        if (ks + 1 < ksteps) {
            const int nb = buf ^ 1;
            *(v4i*)(&xs[nb][r0 * LDS_STRIDE + o0]) = xreg0;
            *(v4i*)(&ws[nb][r0 * LDS_STRIDE + o0]) = wreg0;
            *(v4i*)(&xs[nb][r1 * LDS_STRIDE + o1]) = xreg1;
            *(v4i*)(&ws[nb][r1 * LDS_STRIDE + o1]) = wreg1;
            buf = nb;
            __syncthreads();
        }
#endif
    }

    #pragma unroll
    for (int i = 0; i < 2; ++i) {
        #pragma unroll
        for (int j = 0; j < 4; ++j) {
            const int col = n_blk + wn + j * 16 + lr;
            const int bq  = bias[col];
            #pragma unroll
            for (int e = 0; e < 8; ++e) {
                const int row = m_blk + wm + i * 16 + lh * 8 + e;
                const int v   = acc[i][j][e] + bq;
                if (MODE == 0) {
                    float f = fmaxf((float)v, 0.0f) * 0.005f;
                    int   q = (int)rintf(f);
                    q = q > 127 ? 127 : q;
                    hout[(size_t)row * N + col] = (signed char)q;
                } else if (MODE == 1) {
                    float f = 1e-4f * (float)v + xres[(size_t)row * N + col];
                    fout[(size_t)row * N + col] = f;
                } else {
                    float f = 1e-4f * (float)v;
                    f = fminf(fmaxf(f, -5.0f), 2.0f);
                    fout[(size_t)row * N + col] = f;
                }
            }
        }
    }
}

__global__ __launch_bounds__(256)
void l2norm_kernel(float* __restrict__ out, int D)
{
    __shared__ float red[8];
    __shared__ float scale_s;
    float* p = out + (size_t)blockIdx.x * D;

    float ss = 0.0f;
    for (int i = threadIdx.x; i < D; i += 256) {
        float v = p[i];
        ss += v * v;
    }
    #pragma unroll
    for (int off = 16; off > 0; off >>= 1)
        ss += __shfl_down(ss, off, 32);
    if ((threadIdx.x & 31) == 0) red[threadIdx.x >> 5] = ss;
    __syncthreads();
    if (threadIdx.x == 0) {
        float s = 0.0f;
        #pragma unroll
        for (int i = 0; i < 8; ++i) s += red[i];
        float nrm = fmaxf(sqrtf(s), 1e-12f);
        scale_s = 1.0f / nrm;
    }
    __syncthreads();
    const float sc = scale_s;
    for (int i = threadIdx.x; i < D; i += 256) p[i] *= sc;
}

extern "C" void kernel_launch(void* const* d_in, const int* in_sizes, int n_in,
                              void* d_out, int out_size, void* d_ws, size_t ws_size,
                              hipStream_t stream)
{
    const float* x     = (const float*)d_in[0];
    const int*   w_mu1 = (const int*)d_in[1];
    const int*   b_mu1 = (const int*)d_in[2];
    const int*   w_mu2 = (const int*)d_in[3];
    const int*   b_mu2 = (const int*)d_in[4];
    const int*   w_lv1 = (const int*)d_in[5];
    const int*   b_lv1 = (const int*)d_in[6];
    const int*   w_lv2 = (const int*)d_in[7];
    const int*   b_lv2 = (const int*)d_in[8];

    const int B = 16384, D = 1024, H = 4096;

    char* wsp = (char*)d_ws;
    signed char* xq  = (signed char*)wsp; wsp += (size_t)B * D;
    signed char* w1q = (signed char*)wsp; wsp += (size_t)H * D;
    signed char* w2q = (signed char*)wsp; wsp += (size_t)H * D;
    signed char* hq  = (signed char*)wsp; wsp += (size_t)B * H;

    float* mu_out = (float*)d_out;
    float* lv_out = (float*)d_out + (size_t)B * D;

    const int nx4 = (B * D) / 4;
    const int nw4 = (H * D) / 4;
    dim3 blk(256);
    dim3 g1(B / 128, H / 128);
    dim3 g2(B / 128, D / 128);

    quantize_x_kernel<<<(nx4 + 255) / 256, blk, 0, stream>>>(x, xq, nx4);

    pack_w_kernel<<<(nw4 + 255) / 256, blk, 0, stream>>>(w_mu1, w1q, nw4);
    pack_w_kernel<<<(nw4 + 255) / 256, blk, 0, stream>>>(w_mu2, w2q, nw4);
    gemm_i8_kernel<0><<<g1, blk, 0, stream>>>(xq, w1q, b_mu1, hq, nullptr, nullptr, B, H, D);
    gemm_i8_kernel<1><<<g2, blk, 0, stream>>>(hq, w2q, b_mu2, nullptr, mu_out, x, B, D, H);
    l2norm_kernel<<<B, blk, 0, stream>>>(mu_out, D);

    pack_w_kernel<<<(nw4 + 255) / 256, blk, 0, stream>>>(w_lv1, w1q, nw4);
    pack_w_kernel<<<(nw4 + 255) / 256, blk, 0, stream>>>(w_lv2, w2q, nw4);
    gemm_i8_kernel<0><<<g1, blk, 0, stream>>>(xq, w1q, b_lv1, hq, nullptr, nullptr, B, H, D);
    gemm_i8_kernel<2><<<g2, blk, 0, stream>>>(hq, w2q, b_lv2, nullptr, lv_out, nullptr, B, D, H);
}
